// GatedMatchRNN_85847806313145
// MI455X (gfx1250) — compile-verified, matrix operand order checked by reading, not run on hardware
//
#include <hip/hip_runtime.h>
#include <hip/hip_bf16.h>
#include <math.h>

typedef __attribute__((ext_vector_type(16))) _Float16 v16h;
typedef __attribute__((ext_vector_type(8)))  float    v8f;

__device__ __forceinline__ float sigmf(float x) { return 1.0f / (1.0f + __expf(-x)); }

// CDNA5 hardware transcendental tanh (V_TANH_F32). The trailing v_nop
// satisfies the TRANS-op co-execution hazard (ISA 7.4: one independent
// instruction before the TRANS result is consumed) since the compiler
// cannot see inside the asm block to schedule it.
__device__ __forceinline__ float fast_tanh(float x) {
  float r;
  asm("v_tanh_f32 %0, %1\n\tv_nop" : "=v"(r) : "v"(x));
  return r;
}

// Load one 16-half WMMA fragment (A or B side) for this lane from a row of
// f32 data. ISA 7.12.2 16-bit layout: vector elems 0..7 = K(kbase+0..7),
// elems 8..15 = K(kbase+16..23); kbase = 0 (lanes 0-15) or 8 (lanes 16-31).
__device__ __forceinline__ v16h load_frag_f32(const float* __restrict__ p,
                                              int kc, int kbase) {
  const float4 q0 = *(const float4*)(p + kc + kbase);
  const float4 q1 = *(const float4*)(p + kc + kbase + 4);
  const float4 q2 = *(const float4*)(p + kc + 16 + kbase);
  const float4 q3 = *(const float4*)(p + kc + 16 + kbase + 4);
  v16h f;
  f[0]  = (_Float16)q0.x; f[1]  = (_Float16)q0.y; f[2]  = (_Float16)q0.z; f[3]  = (_Float16)q0.w;
  f[4]  = (_Float16)q1.x; f[5]  = (_Float16)q1.y; f[6]  = (_Float16)q1.z; f[7]  = (_Float16)q1.w;
  f[8]  = (_Float16)q2.x; f[9]  = (_Float16)q2.y; f[10] = (_Float16)q2.z; f[11] = (_Float16)q2.w;
  f[12] = (_Float16)q3.x; f[13] = (_Float16)q3.y; f[14] = (_Float16)q3.z; f[15] = (_Float16)q3.w;
  return f;
}

// ---------------------------------------------------------------------------
// WMMA GEMM, register-blocked: each wave computes a 16x64 output tile
// (1 A fragment reused across 4 B fragments -> 4 v_wmma per K-step).
//   out = epilogue(A @ W^T + bias [+ bias2])
//   A: [M,K] f32 row-major, W: [N,K] f32 row-major (torch Linear weight)
//   MODE 0: store val; MODE 1: store sigmoid(val) * gsrc[m*N+n]  (GLU gate)
// Requires K % 32 == 0, N % 64 == 0, M % 16 == 0.
// ---------------------------------------------------------------------------
template <int MODE, bool HAS_B2>
__global__ void wmma_gemm_kernel(const float* __restrict__ A,
                                 const float* __restrict__ W,
                                 const float* __restrict__ bias,
                                 const float* __restrict__ bias2,
                                 const float* __restrict__ gsrc,
                                 float* __restrict__ out,
                                 int M, int N, int K) {
  const int wave = (int)((blockIdx.x * blockDim.x + threadIdx.x) >> 5);
  const int lane = (int)(threadIdx.x & 31);
  const int tiles_n = N >> 6;                 // 64-wide column groups
  const int tiles   = (M >> 4) * tiles_n;
  if (wave >= tiles) return;                  // wave-uniform; EXEC stays full
  const int tm = wave / tiles_n;
  const int tg = wave - tm * tiles_n;

  const int l15   = lane & 15;
  const int kbase = (lane >> 4) << 3;         // 0 or 8
  const float* Arow = A + (size_t)(tm * 16 + l15) * K;
  const float* W0 = W + (size_t)(tg * 64 + 0  + l15) * K;
  const float* W1 = W + (size_t)(tg * 64 + 16 + l15) * K;
  const float* W2 = W + (size_t)(tg * 64 + 32 + l15) * K;
  const float* W3 = W + (size_t)(tg * 64 + 48 + l15) * K;

  v8f acc0 = {}, acc1 = {}, acc2 = {}, acc3 = {};
  for (int kc = 0; kc < K; kc += 32) {
    const v16h af  = load_frag_f32(Arow, kc, kbase);
    const v16h bf0 = load_frag_f32(W0, kc, kbase);
    const v16h bf1 = load_frag_f32(W1, kc, kbase);
    const v16h bf2 = load_frag_f32(W2, kc, kbase);
    const v16h bf3 = load_frag_f32(W3, kc, kbase);
    acc0 = __builtin_amdgcn_wmma_f32_16x16x32_f16(false, af, false, bf0, (short)0, acc0, false, false);
    acc1 = __builtin_amdgcn_wmma_f32_16x16x32_f16(false, af, false, bf1, (short)0, acc1, false, false);
    acc2 = __builtin_amdgcn_wmma_f32_16x16x32_f16(false, af, false, bf2, (short)0, acc2, false, false);
    acc3 = __builtin_amdgcn_wmma_f32_16x16x32_f16(false, af, false, bf3, (short)0, acc3, false, false);
  }

  // C/D layout: lane<16 -> M = tm*16 + v, lane>=16 -> M = tm*16 + 8 + v
  const int mbase = tm * 16 + ((lane >> 4) << 3);
  const v8f accs[4] = {acc0, acc1, acc2, acc3};
#pragma unroll
  for (int g = 0; g < 4; ++g) {
    const int nout = tg * 64 + g * 16 + l15;
    const float b = bias[nout] + (HAS_B2 ? bias2[nout] : 0.0f);
#pragma unroll
    for (int v = 0; v < 8; ++v) {
      const int m = mbase + v;
      float val = accs[g][v] + b;
      if (MODE == 1) val = sigmf(val) * gsrc[(size_t)m * N + nout];
      out[(size_t)m * N + nout] = val;
    }
  }
}

// ---------------------------------------------------------------------------
// Fused additive attention + online softmax + context + merge construction.
// One wave per (b,t): s_j = V . tanh(xp[b,t] + yp_eff[b,j]) + V_b  (yp_eff has
// Wq_b + Wvp_b folded in), masked, online-softmaxed, ct accumulated in regs.
// Writes merge[b,t] = [ x[b,t] | ct[b,t] ].
// ---------------------------------------------------------------------------
__global__ void attn_merge_kernel(const float* __restrict__ x,
                                  const unsigned char* __restrict__ x_mask,
                                  const float* __restrict__ y,
                                  const unsigned char* __restrict__ y_mask,
                                  const float* __restrict__ xp,
                                  const float* __restrict__ yp,
                                  const float* __restrict__ Vw,
                                  const float* __restrict__ Vb,
                                  float* __restrict__ merge,
                                  int Bn, int L1, int L2, int H) {
  const int row  = blockIdx.x * (blockDim.x >> 5) + ((int)threadIdx.x >> 5);
  const int lane = (int)threadIdx.x & 31;
  if (row >= Bn * L1) return;               // wave-uniform
  const int b  = row / L1;
  const int h0 = lane * 4;                  // 4 h-components per lane (H=128)

  const float4 xq = *(const float4*)(xp + (size_t)row * H + h0);
  const float4 vv = *(const float4*)(Vw + h0);
  const float  vb = Vb[0];
  const float  xm = 1.0f - (float)x_mask[row];

  float m = -1e30f, l = 0.0f;
  float a0 = 0.f, a1 = 0.f, a2 = 0.f, a3 = 0.f;

  for (int j = 0; j < L2; ++j) {
    const size_t yo = ((size_t)b * L2 + j) * H + h0;
    // pull upcoming rows toward the WGP while the tanh chain runs
    __builtin_prefetch(yp + yo + 4 * H, 0, 0);
    __builtin_prefetch(y + yo + 4 * H, 0, 0);
    const float4 yq = *(const float4*)(yp + yo);
    float p = vv.x * fast_tanh(xq.x + yq.x)
            + vv.y * fast_tanh(xq.y + yq.y)
            + vv.z * fast_tanh(xq.z + yq.z)
            + vv.w * fast_tanh(xq.w + yq.w);
#pragma unroll
    for (int off = 16; off > 0; off >>= 1) p += __shfl_xor(p, off, 32);
    float s = p + vb;

    const float ymsk = (float)y_mask[b * L2 + j];
    s = s * (xm * (1.0f - ymsk));
    if (ymsk != 0.0f) s = -1e30f;

    const float mn = fmaxf(m, s);
    const float sc = __expf(m - mn);
    const float e  = __expf(s - mn);
    const float4 yv = *(const float4*)(y + yo);
    l  = l  * sc + e;
    a0 = a0 * sc + e * yv.x;
    a1 = a1 * sc + e * yv.y;
    a2 = a2 * sc + e * yv.z;
    a3 = a3 * sc + e * yv.w;
    m = mn;
  }

  const float inv = 1.0f / l;
  float* mrow = merge + (size_t)row * (2 * H);
  *(float4*)(mrow + h0)     = *(const float4*)(x + (size_t)row * H + h0);
  *(float4*)(mrow + H + h0) = make_float4(a0 * inv, a1 * inv, a2 * inv, a3 * inv);
}

// ---------------------------------------------------------------------------
// Sequential LSTM scan: single workgroup, thread = (batch, hidden-cell).
// gx already holds x-gates + b_ih + b_hh; recurrence adds h @ W_hh^T.
// Hidden state lives in LDS (broadcast reads); cell state in a register.
// ---------------------------------------------------------------------------
__global__ void lstm_kernel(const float* __restrict__ gx,   // [B,L1,4H]
                            const float* __restrict__ Whh,  // [4H,H]
                            float* __restrict__ out,        // [B,L1,H]
                            int Bn, int L1, int H) {
  __shared__ float hsh[8 * 128];
  const int tid = (int)threadIdx.x;       // Bn*H threads (1024)
  const int b   = tid >> 7;               // /H
  const int hi  = tid & 127;
  hsh[tid] = 0.0f;
  float c = 0.0f;
  const float* w0 = Whh + (size_t)(0 * H + hi) * H;
  const float* w1 = Whh + (size_t)(1 * H + hi) * H;
  const float* w2 = Whh + (size_t)(2 * H + hi) * H;
  const float* w3 = Whh + (size_t)(3 * H + hi) * H;
  __syncthreads();

  for (int t = 0; t < L1; ++t) {
    const float* g = gx + (size_t)(b * L1 + t) * (4 * H);
    // prefetch next timestep's gate row
    __builtin_prefetch(g + 4 * H, 0, 0);
    float di = g[hi], df = g[H + hi], dg = g[2 * H + hi], dz = g[3 * H + hi];
    const float* hb = hsh + b * H;
#pragma unroll 8
    for (int k = 0; k < 128; ++k) {
      const float hv = hb[k];
      di += hv * w0[k];
      df += hv * w1[k];
      dg += hv * w2[k];
      dz += hv * w3[k];
    }
    const float cn = sigmf(df) * c + sigmf(di) * fast_tanh(dg);
    const float hn = sigmf(dz) * fast_tanh(cn);
    c = cn;
    __syncthreads();
    hsh[b * H + hi] = hn;
    out[(size_t)(b * L1 + t) * H + hi] = hn;
    __syncthreads();
  }
}

// ---------------------------------------------------------------------------
extern "C" void kernel_launch(void* const* d_in, const int* in_sizes, int n_in,
                              void* d_out, int out_size, void* d_ws, size_t ws_size,
                              hipStream_t stream) {
  (void)in_sizes; (void)n_in; (void)out_size; (void)ws_size;
  const int B = 8, L1 = 384, L2 = 384, H = 128;

  const float*         x      = (const float*)d_in[0];
  const unsigned char* x_mask = (const unsigned char*)d_in[1];
  const float*         y      = (const float*)d_in[2];
  const unsigned char* y_mask = (const unsigned char*)d_in[3];
  const float* Wq_w  = (const float*)d_in[4];
  const float* Wq_b  = (const float*)d_in[5];
  const float* Wup_w = (const float*)d_in[6];
  const float* Wup_b = (const float*)d_in[7];
  /* Wvp_w (d_in[8]) unused: zero hidden state -> projection == bias */
  const float* Wvp_b = (const float*)d_in[9];
  const float* V_w   = (const float*)d_in[10];
  const float* V_b   = (const float*)d_in[11];
  const float* Wg_w  = (const float*)d_in[12];
  const float* Wg_b  = (const float*)d_in[13];
  const float* W_ih  = (const float*)d_in[14];
  const float* b_ih  = (const float*)d_in[15];
  const float* W_hh  = (const float*)d_in[16];
  const float* b_hh  = (const float*)d_in[17];
  float* out = (float*)d_out;

  float* ws      = (float*)d_ws;
  float* xp      = ws;                               // B*L1*H
  float* ypj     = xp      + (size_t)B * L1 * H;     // B*L2*H  (bias = Wq_b+Wvp_b)
  float* merge   = ypj     + (size_t)B * L2 * H;     // B*L1*2H
  float* lstm_in = merge   + (size_t)B * L1 * 2 * H; // B*L1*2H
  float* gxbuf   = lstm_in + (size_t)B * L1 * 2 * H; // B*L1*4H

  const dim3 blk(256);                               // 8 waves / block
  auto blocks_for = [](int M, int N) { return ((M >> 4) * (N >> 6) + 7) / 8; };

  // x_proj = x @ Wup^T + Wup_b
  wmma_gemm_kernel<0, false><<<blocks_for(B * L1, H), blk, 0, stream>>>(
      x, Wup_w, Wup_b, nullptr, nullptr, xp, B * L1, H, H);
  // y_proj_eff = y @ Wq^T + (Wq_b + Wvp_b)
  wmma_gemm_kernel<0, true><<<blocks_for(B * L2, H), blk, 0, stream>>>(
      y, Wq_w, Wq_b, Wvp_b, nullptr, ypj, B * L2, H, H);
  // fused attention scores + online softmax + context + merge=[x|ct]
  attn_merge_kernel<<<(B * L1 + 7) / 8, blk, 0, stream>>>(
      x, x_mask, y, y_mask, xp, ypj, V_w, V_b, merge, B, L1, L2, H);
  // lstm_in = sigmoid(merge @ Wg^T + Wg_b) * merge
  wmma_gemm_kernel<1, false><<<blocks_for(B * L1, 2 * H), blk, 0, stream>>>(
      merge, Wg_w, Wg_b, nullptr, merge, lstm_in, B * L1, 2 * H, 2 * H);
  // gates_x = lstm_in @ W_ih^T + b_ih + b_hh
  wmma_gemm_kernel<0, true><<<blocks_for(B * L1, 4 * H), blk, 0, stream>>>(
      lstm_in, W_ih, b_ih, b_hh, nullptr, gxbuf, B * L1, 4 * H, 2 * H);
  // sequential LSTM scan
  lstm_kernel<<<1, B * H, 0, stream>>>(gxbuf, W_hh, out, B, L1, H);
}
